// CausalSelfAttention_1649267441824
// MI455X (gfx1250) — matrix operand order repaired and checked by reading, not compiled, not run on hardware
//
#include <hip/hip_runtime.h>
#include <hip/hip_bf16.h>

// ---------------------------------------------------------------------------
// CausalSelfAttention on gfx1250 (MI455X): bf16 WMMA + TDM async tensor loads.
//   B=2, S=2048, C=1024, H=16, HD=64
// ---------------------------------------------------------------------------

#define BATCH 2
#define SEQ   2048
#define EMBD  1024
#define NHEAD 16
#define HDIM  64
#define MROWS (BATCH * SEQ)          // 4096
#define QKVC  (3 * EMBD)             // 3072

typedef __attribute__((ext_vector_type(16))) __bf16        v16bf;
typedef __attribute__((ext_vector_type(8)))  float         v8f;
typedef __attribute__((ext_vector_type(8)))  unsigned int  v8u;
typedef __attribute__((ext_vector_type(4)))  unsigned int  u32x4;
typedef __attribute__((ext_vector_type(8)))  int           i32x8;
typedef __attribute__((ext_vector_type(4)))  int           i32x4;

// --------------------------- TDM availability ------------------------------
#if defined(__has_builtin)
#if __has_builtin(__builtin_amdgcn_tensor_load_to_lds) && \
    __has_builtin(__builtin_amdgcn_s_wait_tensorcnt)
#define USE_TDM 1
#endif
#endif
#ifndef USE_TDM
#define USE_TDM 0
#endif

__device__ __forceinline__ unsigned short f2bf(float f) {
    unsigned int u = __float_as_uint(f);
    u += 0x7FFFu + ((u >> 16) & 1u);      // round-to-nearest-even
    return (unsigned short)(u >> 16);
}

__device__ __forceinline__ v16bf frag16(uint4 q0, uint4 q1) {
    v8u r = { q0.x, q0.y, q0.z, q0.w, q1.x, q1.y, q1.z, q1.w };
    return __builtin_bit_cast(v16bf, r);
}

// A-fragment (16x32 bf16): dwords at k0+hi*8 and k0+16+hi*8 within the row.
__device__ __forceinline__ v16bf load_afrag(const unsigned short* base, int k0, int hi) {
    const uint4* p = (const uint4*)(base + k0 + hi * 8);
    return frag16(p[0], p[2]);
}

// B-fragment (32x16 bf16) with B[k][n] = W[n][k]: dwords at k0+hi*8 and k0+16+hi*8.
__device__ __forceinline__ v16bf load_bfrag(const unsigned short* base, int k0, int hi) {
    const uint4* p = (const uint4*)(base + k0 + hi * 8);
    return frag16(p[0], p[2]);
}

__device__ __forceinline__ v8f wmma_bf16(v16bf a, v16bf b, v8f c) {
    return __builtin_amdgcn_wmma_f32_16x16x32_bf16(
        false, a, false, b, (short)0, c, false, false);
}

__device__ __forceinline__ float hmax16(float v) {
    v = fmaxf(v, __shfl_xor(v, 1, 32));
    v = fmaxf(v, __shfl_xor(v, 2, 32));
    v = fmaxf(v, __shfl_xor(v, 4, 32));
    v = fmaxf(v, __shfl_xor(v, 8, 32));
    return v;
}
__device__ __forceinline__ float hsum16(float v) {
    v += __shfl_xor(v, 1, 32);
    v += __shfl_xor(v, 2, 32);
    v += __shfl_xor(v, 4, 32);
    v += __shfl_xor(v, 8, 32);
    return v;
}

#if USE_TDM
// Issue a 2D TDM tile load: tile_rows x tile_k bf16 elements, row stride
// row_stride (elements), packed contiguously into LDS at lds_off.
// D# layout per CDNA5 ISA ch.8 (group0 128b, group1 256b; groups 2/3 zero for 2D).
__device__ __forceinline__ void tdm_load_2d(const unsigned short* gaddr,
                                            unsigned lds_off,
                                            int tile_k, int tile_rows,
                                            int row_stride,
                                            unsigned t_dim0, unsigned t_dim1) {
    unsigned long long ga = (unsigned long long)(uintptr_t)gaddr;
    u32x4 g0;
    g0[0] = 1u;                                             // count=1, user mode
    g0[1] = lds_off;                                        // lds_addr
    g0[2] = (unsigned)(ga & 0xFFFFFFFFull);                 // global_addr[31:0]
    g0[3] = (unsigned)((ga >> 32) & 0x01FFFFFFull)          // global_addr[56:32]
          | (2u << 30);                                     // type = 2 ("image")
    i32x8 g1;
    g1[0] = (int)(1u << 16);                                // data_size=1 -> 2 bytes
    g1[1] = (int)((t_dim0 & 0xFFFFu) << 16);                // tensor_dim0[15:0]
    g1[2] = (int)(((t_dim0 >> 16) & 0xFFFFu)                // tensor_dim0[31:16]
          | ((t_dim1 & 0xFFFFu) << 16));                    // tensor_dim1[15:0]
    g1[3] = (int)(((t_dim1 >> 16) & 0xFFFFu)                // tensor_dim1[31:16]
          | (((unsigned)tile_k & 0xFFFFu) << 16));          // tile_dim0
    g1[4] = (int)((unsigned)tile_rows & 0xFFFFu);           // tile_dim1 (tile_dim2=0)
    g1[5] = (int)(unsigned)row_stride;                      // tensor_dim0_stride[31:0]
    g1[6] = 0;                                              // stride[47:32], dim1_stride lo
    g1[7] = 0;                                              // dim1_stride hi (unused, 2D)
    i32x4 z4 = {0, 0, 0, 0};
#if __clang_major__ >= 23
    i32x8 z8 = {0, 0, 0, 0, 0, 0, 0, 0};
    __builtin_amdgcn_tensor_load_to_lds(g0, g1, z4, z4, z8, 0);
#else
    __builtin_amdgcn_tensor_load_to_lds(g0, g1, z4, z4, 0);
#endif
}
#endif // USE_TDM

// ---------------------------------------------------------------------------
// fp32 -> bf16 conversion, 4 elems / thread
// ---------------------------------------------------------------------------
__global__ __launch_bounds__(256) void cvt_f32_bf16(const float4* __restrict__ in,
                                                    uint2* __restrict__ out, int n4) {
    int i = blockIdx.x * 256 + threadIdx.x;
    if (i < n4) {
        float4 f = in[i];
        uint2 o;
        o.x = (unsigned)f2bf(f.x) | ((unsigned)f2bf(f.y) << 16);
        o.y = (unsigned)f2bf(f.z) | ((unsigned)f2bf(f.w) << 16);
        out[i] = o;
    }
}

// ---------------------------------------------------------------------------
// GEMM: Out[M,N] = A[M,K] * W[N,K]^T  (bf16 in, f32 acc)
// Block tile 64x256 (8 waves, each 32x64). TDM stages A(64x32) and B(256x32)
// k-slices into double-buffered LDS; waves consume via ds_load_b128 fragments.
// ---------------------------------------------------------------------------
template <bool OUT_BF16>
__global__ __launch_bounds__(256) void gemm_wmma(const unsigned short* __restrict__ A,
                                                 const unsigned short* __restrict__ W,
                                                 void* __restrict__ Out,
                                                 int M, int N, int K) {
    const int lane = threadIdx.x & 31;
    const int w    = threadIdx.x >> 5;
    const int hi   = lane >> 4;
    const int lc   = lane & 15;

    const int blkM = blockIdx.x * 64;
    const int blkN = blockIdx.y * 256;
    const int m0w  = (w & 1) * 32;       // wave offset inside block tile
    const int n0w  = (w >> 1) * 64;

    v8f acc[2][4] = {};

#if USE_TDM
    __shared__ __align__(16) unsigned short ldsA[2][64 * 32];    //  8 KB
    __shared__ __align__(16) unsigned short ldsB[2][256 * 32];   // 32 KB

    const unsigned offA0 = (unsigned)(uintptr_t)(const void*)&ldsA[0][0];
    const unsigned offA1 = (unsigned)(uintptr_t)(const void*)&ldsA[1][0];
    const unsigned offB0 = (unsigned)(uintptr_t)(const void*)&ldsB[0][0];
    const unsigned offB1 = (unsigned)(uintptr_t)(const void*)&ldsB[1][0];

    const unsigned short* Ablk = A + (size_t)blkM * K;
    const unsigned short* Wblk = W + (size_t)blkN * K;

    if (w == 0) {   // prologue: prime both buffers (4 TDM ops in flight)
        tdm_load_2d(Ablk + 0,  offA0, 32, 64,  K, (unsigned)K, 64u);
        tdm_load_2d(Wblk + 0,  offB0, 32, 256, K, (unsigned)K, 256u);
        tdm_load_2d(Ablk + 32, offA1, 32, 64,  K, (unsigned)K, 64u);
        tdm_load_2d(Wblk + 32, offB1, 32, 256, K, (unsigned)K, 256u);
    }

    for (int k0 = 0; k0 < K; k0 += 32) {
        const int buf = (k0 >> 5) & 1;
        if (w == 0) {   // oldest pair (current buffer) must be complete
            if (k0 + 32 < K) __builtin_amdgcn_s_wait_tensorcnt(2);
            else             __builtin_amdgcn_s_wait_tensorcnt(0);
        }
        __syncthreads();

        v16bf a0 = load_afrag(&ldsA[buf][(m0w + lc) * 32], 0, hi);
        v16bf a1 = load_afrag(&ldsA[buf][(m0w + 16 + lc) * 32], 0, hi);
#pragma unroll
        for (int j = 0; j < 4; ++j) {
            v16bf b = load_bfrag(&ldsB[buf][(n0w + j * 16 + lc) * 32], 0, hi);
            acc[0][j] = wmma_bf16(a0, b, acc[0][j]);
            acc[1][j] = wmma_bf16(a1, b, acc[1][j]);
        }

        __syncthreads();    // everyone done reading `buf` before refill
        if (w == 0 && k0 + 64 < K) {
            const int kn = k0 + 64;
            tdm_load_2d(Ablk + kn, buf ? offA1 : offA0, 32, 64,  K, (unsigned)K, 64u);
            tdm_load_2d(Wblk + kn, buf ? offB1 : offB0, 32, 256, K, (unsigned)K, 256u);
        }
    }
#else
    // Fallback: direct global fragment streaming (L2 resident anyway).
    const unsigned short* arow0 = A + (size_t)(blkM + m0w + lc) * K;
    const unsigned short* arow1 = A + (size_t)(blkM + m0w + 16 + lc) * K;
    const unsigned short* wrow[4];
#pragma unroll
    for (int j = 0; j < 4; ++j) wrow[j] = W + (size_t)(blkN + n0w + j * 16 + lc) * K;

    for (int k0 = 0; k0 < K; k0 += 32) {
        v16bf a0 = load_afrag(arow0, k0, hi);
        v16bf a1 = load_afrag(arow1, k0, hi);
#pragma unroll
        for (int j = 0; j < 4; ++j) {
            v16bf b = load_bfrag(wrow[j], k0, hi);
            acc[0][j] = wmma_bf16(a0, b, acc[0][j]);
            acc[1][j] = wmma_bf16(a1, b, acc[1][j]);
        }
    }
#endif

#pragma unroll
    for (int i = 0; i < 2; ++i)
#pragma unroll
        for (int j = 0; j < 4; ++j)
#pragma unroll
            for (int r = 0; r < 8; ++r) {
                int mm = blkM + m0w + i * 16 + r + hi * 8;
                int nn = blkN + n0w + j * 16 + lc;
                if (OUT_BF16)
                    ((unsigned short*)Out)[(size_t)mm * N + nn] = f2bf(acc[i][j][r]);
                else
                    ((float*)Out)[(size_t)mm * N + nn] = acc[i][j][r];
            }
}

// ---------------------------------------------------------------------------
// Repack V from qkv[(b*S+s)][2*EMBD + h*64 + d] -> vt[(b*H+h)*64 + d][s]
// ---------------------------------------------------------------------------
__global__ __launch_bounds__(256) void repack_v(const unsigned short* __restrict__ qkv,
                                                unsigned short* __restrict__ vt) {
    int i  = blockIdx.x * 256 + threadIdx.x;        // over B*H*64*S = 4M
    int s  = i & (SEQ - 1);
    int d  = (i >> 11) & (HDIM - 1);
    int bh = i >> 17;                               // 0..31
    int b  = bh >> 4, h = bh & 15;
    vt[i] = qkv[(size_t)(b * SEQ + s) * QKVC + 2 * EMBD + h * HDIM + d];
}

// ---------------------------------------------------------------------------
// Flash-style causal attention. One wave per 16-query tile of one (b,h).
// ---------------------------------------------------------------------------
__global__ __launch_bounds__(256) void attn_wmma(const unsigned short* __restrict__ qkv,
                                                 const unsigned short* __restrict__ vt,
                                                 unsigned short* __restrict__ y) {
    __shared__ __align__(16) unsigned short plds[8][16 * 32];   // 1KB / wave

    const int lane = threadIdx.x & 31;
    const int w    = threadIdx.x >> 5;
    const int hi   = lane >> 4;
    const int lc   = lane & 15;

    const int gw = blockIdx.x * 8 + w;
    const int qt = gw & 127;                // query tile (S/16 = 128)
    const int bh = gw >> 7;
    const int b  = bh >> 4, h = bh & 15;

    const unsigned short* qrow = qkv + (size_t)(b * SEQ + qt * 16 + lc) * QKVC + h * HDIM;
    v16bf qa0 = load_afrag(qrow, 0, hi);
    v16bf qa1 = load_afrag(qrow, 32, hi);

    float mrun[8], lrun[8];
#pragma unroll
    for (int r = 0; r < 8; ++r) { mrun[r] = -3.0e38f; lrun[r] = 0.f; }
    v8f o[4] = {};

    const unsigned short* vbase = vt + (size_t)bh * HDIM * SEQ;
    const int kend = qt * 16 + 16;
    const float scale = 0.125f;

    for (int kb = 0; kb < kend; kb += 32) {
        v8f s0 = {}, s1 = {};
        {
            const unsigned short* kr0 = qkv + (size_t)(b * SEQ + kb + lc) * QKVC + EMBD + h * HDIM;
            const unsigned short* kr1 = kr0 + (size_t)16 * QKVC;
            s0 = wmma_bf16(qa0, load_bfrag(kr0, 0, hi), s0);
            s0 = wmma_bf16(qa1, load_bfrag(kr0, 32, hi), s0);
            s1 = wmma_bf16(qa0, load_bfrag(kr1, 0, hi), s1);
            s1 = wmma_bf16(qa1, load_bfrag(kr1, 32, hi), s1);
        }

#pragma unroll
        for (int r = 0; r < 8; ++r) {
            const int q = qt * 16 + hi * 8 + r;
            float x0 = s0[r] * scale; if (kb + lc > q)      x0 = -3.0e38f;
            float x1 = s1[r] * scale; if (kb + 16 + lc > q) x1 = -3.0e38f;
            float tmax = hmax16(fmaxf(x0, x1));
            float mnew = fmaxf(mrun[r], tmax);
            float alpha = __expf(mrun[r] - mnew);
            float p0 = __expf(x0 - mnew);
            float p1 = __expf(x1 - mnew);
            lrun[r] = lrun[r] * alpha + hsum16(p0 + p1);
            mrun[r] = mnew;
#pragma unroll
            for (int nt = 0; nt < 4; ++nt) o[nt][r] *= alpha;
            plds[w][(r + hi * 8) * 32 + lc]      = f2bf(p0);
            plds[w][(r + hi * 8) * 32 + 16 + lc] = f2bf(p1);
        }
        asm volatile("" ::: "memory");   // keep DS store->load order (HW in-order per wave)

        v16bf pa = load_afrag(&plds[w][0] + (size_t)lc * 32, 0, hi);

#pragma unroll
        for (int nt = 0; nt < 4; ++nt) {
            const unsigned short* vrow = vbase + (size_t)(nt * 16 + lc) * SEQ + kb;
            o[nt] = wmma_bf16(pa, load_bfrag(vrow, 0, hi), o[nt]);
        }
    }

#pragma unroll
    for (int r = 0; r < 8; ++r) {
        float inv = 1.0f / lrun[r];
        int q = qt * 16 + r + hi * 8;
        unsigned short* yrow = y + (size_t)(b * SEQ + q) * EMBD + h * HDIM;
#pragma unroll
        for (int nt = 0; nt < 4; ++nt)
            yrow[nt * 16 + lc] = f2bf(o[nt][r] * inv);
    }
}

// ---------------------------------------------------------------------------
// Launch
// ---------------------------------------------------------------------------
extern "C" void kernel_launch(void* const* d_in, const int* in_sizes, int n_in,
                              void* d_out, int out_size, void* d_ws, size_t ws_size,
                              hipStream_t stream) {
    const float* x      = (const float*)d_in[0];   // [2,2048,1024]
    const float* w_qkv  = (const float*)d_in[1];   // [3072,1024]
    const float* w_proj = (const float*)d_in[2];   // [1024,1024]
    float* out = (float*)d_out;                    // f32

    unsigned short* ws     = (unsigned short*)d_ws;
    unsigned short* xb     = ws;                                          //  4M
    unsigned short* wqkvb  = xb     + (size_t)MROWS * EMBD;               //  3M
    unsigned short* wprojb = wqkvb  + (size_t)QKVC * EMBD;                //  1M
    unsigned short* qkvb   = wprojb + (size_t)EMBD * EMBD;                // 12M
    unsigned short* vtb    = qkvb   + (size_t)MROWS * QKVC;               //  4M
    unsigned short* yb     = vtb    + (size_t)BATCH * NHEAD * HDIM * SEQ; //  4M

    {
        int n4;
        n4 = (MROWS * EMBD) / 4;
        cvt_f32_bf16<<<n4 / 256, 256, 0, stream>>>((const float4*)x, (uint2*)xb, n4);
        n4 = (QKVC * EMBD) / 4;
        cvt_f32_bf16<<<n4 / 256, 256, 0, stream>>>((const float4*)w_qkv, (uint2*)wqkvb, n4);
        n4 = (EMBD * EMBD) / 4;
        cvt_f32_bf16<<<n4 / 256, 256, 0, stream>>>((const float4*)w_proj, (uint2*)wprojb, n4);
    }

    gemm_wmma<true><<<dim3(MROWS / 64, QKVC / 256), 256, 0, stream>>>(
        xb, wqkvb, qkvb, MROWS, QKVC, EMBD);

    repack_v<<<(BATCH * NHEAD * HDIM * SEQ) / 256, 256, 0, stream>>>(qkvb, vtb);

    attn_wmma<<<(BATCH * NHEAD * (SEQ / 16)) / 8, 256, 0, stream>>>(qkvb, vtb, yb);

    gemm_wmma<false><<<dim3(MROWS / 64, EMBD / 256), 256, 0, stream>>>(
        yb, wprojb, out, MROWS, EMBD, EMBD);
}
